// RecurrentBlock_11338713662141
// MI455X (gfx1250) — matrix operand order repaired and checked by reading, not compiled, not run on hardware
//
#include <hip/hip_runtime.h>
#include <hip/hip_bf16.h>

typedef __bf16 bf16;
typedef __attribute__((ext_vector_type(16))) __bf16 v16bf;
typedef __attribute__((ext_vector_type(8)))  float  v8f;

// Fixed problem dims from reference setup_inputs()
constexpr int B_  = 2;
constexpr int S_  = 4096;
constexpr int H_  = 2048;
constexpr int L_  = 2048;
constexpr int NH_ = 16;
constexpr int BW_ = 128;
constexpr int KW_ = 4;
constexpr int M_  = B_ * S_;   // 8192 GEMM rows

// ---------------------------------------------------------------------------
// WMMA fragment helpers (CDNA5 16x16x32 bf16, wave32)
// A 16x32: lane<16 row=lane holds K {0..7} in elems 0..7, K {16..23} in 8..15;
//          lane>=16 row=lane-16 holds K {8..15}, {24..31}.
// B stored N-major (bt[n][k]): lane<16 col=lane reads k rows {0..7},{16..23},
//          lane>=16 col=lane-16 reads k rows {8..15},{24..31}.
// C/D v8f: elem i -> M = ((lane>=16)?8:0)+i, N = lane&15.
// ---------------------------------------------------------------------------
__device__ inline v16bf load_a_frag(const bf16* __restrict__ a, int lda, int lane) {
  const int row = lane & 15;
  const int kh  = lane >> 4;
  const bf16* p = a + row * lda + kh * 8;
  v16bf r;
  ((uint4*)&r)[0] = *(const uint4*)(p);
  ((uint4*)&r)[1] = *(const uint4*)(p + 16);
  return r;
}
__device__ inline v16bf load_b_frag(const bf16* __restrict__ bt, int ldb, int lane) {
  const int col = lane & 15;
  const int kh  = lane >> 4;
  const bf16* p = bt + col * ldb + kh * 8;
  v16bf r;
  ((uint4*)&r)[0] = *(const uint4*)(p);
  ((uint4*)&r)[1] = *(const uint4*)(p + 16);
  return r;
}
__device__ inline v8f wmma_bf16(v16bf a, v16bf b, v8f c) {
  return __builtin_amdgcn_wmma_f32_16x16x32_bf16(false, a, false, b, (short)0, c,
                                                 false, false);
}

__device__ inline float gelu_tanh(float x) {
  const float c = 0.7978845608028654f;   // sqrt(2/pi)
  float t = tanhf(c * (x + 0.044715f * x * x * x));
  return 0.5f * x * (1.f + t);
}
__device__ inline float sigmoidf_(float x) { return 1.f / (1.f + expf(-x)); }

// ---------------------------------------------------------------------------
// Generic bf16-compute / f32-accumulate GEMM:  C[MxN] = epi(A[MxK] * B[KxN] + bias)
// Block tile 128x128, BK=32, 256 threads (8 waves: 2x4), wave tile 64x32.
// MODE: 0 = bias, 1 = bias+gelu
// ---------------------------------------------------------------------------
template <int MODE>
__global__ __launch_bounds__(256) void gemm_bf16_kernel(
    const float* __restrict__ A, const float* __restrict__ Bm,
    const float* __restrict__ bias, float* __restrict__ C,
    int M, int N, int K) {
  constexpr int LS = 40;  // LDS row stride (bf16 elems): 80B, 16B-aligned, conflict-free
  __shared__ __align__(16) bf16 sA[128 * LS];
  __shared__ __align__(16) bf16 sB[128 * LS];

  const int tid  = threadIdx.x;
  const int lane = tid & 31;
  const int wid  = tid >> 5;
  const int wm   = wid >> 2;  // 0..1
  const int wn   = wid & 3;   // 0..3
  const int m0   = blockIdx.y * 128;
  const int n0   = blockIdx.x * 128;

  v8f acc[4][2];
#pragma unroll
  for (int i = 0; i < 4; ++i)
#pragma unroll
    for (int j = 0; j < 2; ++j) acc[i][j] = (v8f)(0.f);

  for (int k0 = 0; k0 < K; k0 += 32) {
    // Stage A tile (128x32 f32 -> bf16, row-major)
#pragma unroll
    for (int i = 0; i < 4; ++i) {
      int idx = tid + i * 256;          // 1024 float4 quads
      int row = idx >> 3;               // 8 quads per 32-elem row
      int kq  = (idx & 7) << 2;
      float4 v = *(const float4*)(A + (size_t)(m0 + row) * K + k0 + kq);
      bf16* sp = &sA[row * LS + kq];
      sp[0] = (bf16)v.x; sp[1] = (bf16)v.y; sp[2] = (bf16)v.z; sp[3] = (bf16)v.w;
    }
    // Stage B tile (32x128 f32 -> bf16, stored N-major/transposed)
#pragma unroll
    for (int i = 0; i < 4; ++i) {
      int idx  = tid + i * 256;
      int krow = idx >> 5;              // 32 quads per 128-col k-row
      int nq   = (idx & 31) << 2;
      float4 v = *(const float4*)(Bm + (size_t)(k0 + krow) * N + n0 + nq);
      sB[(nq + 0) * LS + krow] = (bf16)v.x;
      sB[(nq + 1) * LS + krow] = (bf16)v.y;
      sB[(nq + 2) * LS + krow] = (bf16)v.z;
      sB[(nq + 3) * LS + krow] = (bf16)v.w;
    }
    __syncthreads();

    v16bf af[4], bfm[2];
#pragma unroll
    for (int im = 0; im < 4; ++im)
      af[im] = load_a_frag(&sA[(wm * 64 + im * 16) * LS], LS, lane);
#pragma unroll
    for (int in = 0; in < 2; ++in)
      bfm[in] = load_b_frag(&sB[(wn * 32 + in * 16) * LS], LS, lane);
#pragma unroll
    for (int im = 0; im < 4; ++im)
#pragma unroll
      for (int in = 0; in < 2; ++in)
        acc[im][in] = wmma_bf16(af[im], bfm[in], acc[im][in]);
    __syncthreads();
  }

  // Epilogue
  const int colL = lane & 15;
  const int rb   = (lane >> 4) * 8;
#pragma unroll
  for (int im = 0; im < 4; ++im) {
#pragma unroll
    for (int in = 0; in < 2; ++in) {
      const int mt = m0 + wm * 64 + im * 16 + rb;
      const int nt = n0 + wn * 32 + in * 16 + colL;
      const float bv = bias[nt];
#pragma unroll
      for (int i = 0; i < 8; ++i) {
        float v = acc[im][in][i] + bv;
        if (MODE == 1) v = gelu_tanh(v);
        C[(size_t)(mt + i) * N + nt] = v;
      }
    }
  }
}

// ---------------------------------------------------------------------------
// Causal depthwise conv1d, K=4, left pad 3:  out[b,s,l] = b[l] + sum_k w[l,k]*x[b,s-3+k,l]
// ---------------------------------------------------------------------------
__global__ __launch_bounds__(256) void conv_kernel(
    const float* __restrict__ x, const float* __restrict__ w,
    const float* __restrict__ bias, float* __restrict__ out) {
  size_t i = (size_t)blockIdx.x * 256 + threadIdx.x;  // < B*S*L
  int l = (int)(i % L_);
  int s = (int)((i / L_) % S_);
  float acc = bias[l];
#pragma unroll
  for (int k = 0; k < KW_; ++k) {
    int sp = s - (KW_ - 1) + k;
    if (sp >= 0) acc = fmaf(w[l * KW_ + k], x[i + (size_t)(sp - s) * L_], acc);
  }
  out[i] = acc;
}

// ---------------------------------------------------------------------------
// RG-LRU gates: per-head block-diag GEMMs (ig, rg) via WMMA, fused elementwise
// epilogue producing a_eff and normed.
// Block tile 64 tokens x 128 (=BW) channels, K=128, 256 threads (waves 2x4,
// each wave 32x32 = 2x2 tiles, two accumulator sets).
// ---------------------------------------------------------------------------
__global__ __launch_bounds__(256) void gates_kernel(
    const float* __restrict__ xc, const float* __restrict__ igw,
    const float* __restrict__ igb, const float* __restrict__ rgw,
    const float* __restrict__ rgb, const float* __restrict__ rec,
    const int* __restrict__ pos, float* __restrict__ a_eff,
    float* __restrict__ normed) {
  constexpr int LS = 40;
  __shared__ __align__(16) bf16 sA[64 * LS];
  __shared__ __align__(16) bf16 sI[128 * LS];
  __shared__ __align__(16) bf16 sR[128 * LS];

  const int head = blockIdx.x;
  const int m0   = blockIdx.y * 64;
  const int tid  = threadIdx.x;
  const int lane = tid & 31;
  const int wid  = tid >> 5;
  const int wm   = wid >> 2;  // 0..1
  const int wn   = wid & 3;   // 0..3
  const float* wI = igw + (size_t)head * BW_ * BW_;
  const float* wR = rgw + (size_t)head * BW_ * BW_;

  v8f aI[2][2], aR[2][2];
#pragma unroll
  for (int i = 0; i < 2; ++i)
#pragma unroll
    for (int j = 0; j < 2; ++j) { aI[i][j] = (v8f)(0.f); aR[i][j] = (v8f)(0.f); }

  for (int k0 = 0; k0 < BW_; k0 += 32) {
    // A tile: 64 tokens x 32 k (xc columns head*BW + k0 ..)
#pragma unroll
    for (int i = 0; i < 2; ++i) {
      int idx = tid + i * 256;          // 512 quads
      int row = idx >> 3;
      int kq  = (idx & 7) << 2;
      float4 v = *(const float4*)(xc + (size_t)(m0 + row) * L_ + head * BW_ + k0 + kq);
      bf16* sp = &sA[row * LS + kq];
      sp[0] = (bf16)v.x; sp[1] = (bf16)v.y; sp[2] = (bf16)v.z; sp[3] = (bf16)v.w;
    }
    // Weight tiles: 32 k-rows x 128 cols, stored N-major
#pragma unroll
    for (int i = 0; i < 4; ++i) {
      int idx  = tid + i * 256;
      int krow = idx >> 5;
      int nq   = (idx & 31) << 2;
      float4 v = *(const float4*)(wI + (size_t)(k0 + krow) * BW_ + nq);
      sI[(nq + 0) * LS + krow] = (bf16)v.x;
      sI[(nq + 1) * LS + krow] = (bf16)v.y;
      sI[(nq + 2) * LS + krow] = (bf16)v.z;
      sI[(nq + 3) * LS + krow] = (bf16)v.w;
      float4 u = *(const float4*)(wR + (size_t)(k0 + krow) * BW_ + nq);
      sR[(nq + 0) * LS + krow] = (bf16)u.x;
      sR[(nq + 1) * LS + krow] = (bf16)u.y;
      sR[(nq + 2) * LS + krow] = (bf16)u.z;
      sR[(nq + 3) * LS + krow] = (bf16)u.w;
    }
    __syncthreads();

    v16bf af[2], bi[2], br[2];
#pragma unroll
    for (int im = 0; im < 2; ++im)
      af[im] = load_a_frag(&sA[(wm * 32 + im * 16) * LS], LS, lane);
#pragma unroll
    for (int in = 0; in < 2; ++in) {
      bi[in] = load_b_frag(&sI[(wn * 32 + in * 16) * LS], LS, lane);
      br[in] = load_b_frag(&sR[(wn * 32 + in * 16) * LS], LS, lane);
    }
#pragma unroll
    for (int im = 0; im < 2; ++im)
#pragma unroll
      for (int in = 0; in < 2; ++in) {
        aI[im][in] = wmma_bf16(af[im], bi[in], aI[im][in]);
        aR[im][in] = wmma_bf16(af[im], br[in], aR[im][in]);
      }
    __syncthreads();
  }

  // Fused RG-LRU elementwise epilogue
  const int colL = lane & 15;
  const int rb   = (lane >> 4) * 8;
#pragma unroll
  for (int im = 0; im < 2; ++im) {
#pragma unroll
    for (int in = 0; in < 2; ++in) {
      const int mt = m0 + wm * 32 + im * 16 + rb;
      const int e  = wn * 32 + in * 16 + colL;
      const int l  = head * BW_ + e;
      const float bI = igb[l];
      const float bR = rgb[l];
      const float sp = log1pf(expf(rec[l]));  // softplus(rec_param)
#pragma unroll
      for (int i = 0; i < 8; ++i) {
        const int r = mt + i;                 // flat token index b*S+s
        float ig = sigmoidf_(aI[im][in][i] + bI);
        float rg = sigmoidf_(aR[im][in][i] + bR);
        float log_a = -8.f * rg * sp;
        float a = expf(log_a);
        float mult = sqrtf(1.f - expf(2.f * log_a));  // sqrt_bound fwd
        bool reset = (pos[r] == 0);
        float xv = xc[(size_t)r * L_ + l];
        a_eff[(size_t)r * L_ + l]  = reset ? 0.f : a;
        normed[(size_t)r * L_ + l] = xv * ig * (reset ? 1.f : mult);
      }
    }
  }
}

// ---------------------------------------------------------------------------
// Linear recurrence h = a*h + x over S, fused with hs*y.  One thread per (b,l)
// channel; consecutive threads hit consecutive l -> coalesced per step.
// ---------------------------------------------------------------------------
__global__ __launch_bounds__(256) void scan_kernel(
    const float* __restrict__ a_eff, const float* __restrict__ normed,
    const float* __restrict__ y, float* __restrict__ hsy) {
  const int c = blockIdx.x * 256 + threadIdx.x;  // 0..B*L-1
  const int b = c / L_;
  const int l = c % L_;
  const size_t base = (size_t)b * S_ * L_ + l;
  float h = 0.f;
#pragma unroll 4
  for (int s = 0; s < S_; ++s) {
    const size_t off = base + (size_t)s * L_;
    h = fmaf(a_eff[off], h, normed[off]);
    hsy[off] = h * y[off];
  }
}

// ---------------------------------------------------------------------------
// Host launcher
// ---------------------------------------------------------------------------
extern "C" void kernel_launch(void* const* d_in, const int* in_sizes, int n_in,
                              void* d_out, int out_size, void* d_ws, size_t ws_size,
                              hipStream_t stream) {
  (void)in_sizes; (void)n_in; (void)out_size; (void)ws_size;

  const float* x_in   = (const float*)d_in[0];
  const float* Wy     = (const float*)d_in[1];
  const float* by     = (const float*)d_in[2];
  const float* Wx     = (const float*)d_in[3];
  const float* bx     = (const float*)d_in[4];
  const float* conv_w = (const float*)d_in[5];
  const float* conv_b = (const float*)d_in[6];
  const float* rec    = (const float*)d_in[7];
  const float* igw    = (const float*)d_in[8];
  const float* igb    = (const float*)d_in[9];
  const float* rgw    = (const float*)d_in[10];
  const float* rgb    = (const float*)d_in[11];
  const float* Wout   = (const float*)d_in[12];
  const float* bout   = (const float*)d_in[13];
  const int*   pos    = (const int*)d_in[14];
  float* out = (float*)d_out;

  const size_t tsz = (size_t)M_ * L_ * sizeof(float);  // 64 MB
  char* ws = (char*)d_ws;
  float* bufY  = (float*)(ws);            // y = gelu(X Wy + by)
  float* bufX  = (float*)(ws + tsz);      // xlin -> a_eff (reused)
  float* bufC  = (float*)(ws + 2 * tsz);  // xconv -> hsy (reused)
  float* bufN  = (float*)(ws + 3 * tsz);  // normed

  const dim3 blk(256);
  const dim3 gGemm(L_ / 128, M_ / 128);   // (16, 64)

  // 1) y = gelu(X @ Wy + by)
  gemm_bf16_kernel<1><<<gGemm, blk, 0, stream>>>(x_in, Wy, by, bufY, M_, L_, H_);
  // 2) xlin = X @ Wx + bx
  gemm_bf16_kernel<0><<<gGemm, blk, 0, stream>>>(x_in, Wx, bx, bufX, M_, L_, H_);
  // 3) causal depthwise conv
  conv_kernel<<<dim3((unsigned)((size_t)M_ * L_ / 256)), blk, 0, stream>>>(
      bufX, conv_w, conv_b, bufC);
  // 4) RG-LRU gates (block-diag WMMA GEMMs + fused elementwise)
  gates_kernel<<<dim3(NH_, M_ / 64), blk, 0, stream>>>(
      bufC, igw, igb, rgw, rgb, rec, pos, bufX, bufN);
  // 5) linear recurrence fused with *y  (hsy overwrites xconv buffer)
  scan_kernel<<<dim3((B_ * L_) / 256), blk, 0, stream>>>(bufX, bufN, bufY, bufC);
  // 6) out = hsy @ Wout + bout
  gemm_bf16_kernel<0><<<dim3(H_ / 128, M_ / 128), blk, 0, stream>>>(
      bufC, Wout, bout, out, M_, H_, L_);
}
